// TransformerMessage_88656714925213
// MI455X (gfx1250) — matrix operand order repaired and checked by reading, not compiled, not run on hardware
//
#include <hip/hip_runtime.h>
#include <hip/hip_bf16.h>

#define E_NODE_DIM 64
#define E_EDGE_DIM 32
#define E_HIDDEN   192
#define E_INDIM    160            // 2*64 + 32
#define E_NNODES   100000
#define TILE_E     32             // edges per block tile
#define WIN_STRIDE 164            // padded k-stride for W_in LDS (bf16 units)
#define HW_STRIDE  196            // padded stride for 192-wide rows

typedef __attribute__((ext_vector_type(16))) __bf16 v16bf;
typedef __attribute__((ext_vector_type(8)))  float  v8f;

__device__ __forceinline__ unsigned short f2bf(float f) {
    unsigned u = __float_as_uint(f);
    u += 0x7FFFu + ((u >> 16) & 1u);          // round-to-nearest-even
    return (unsigned short)(u >> 16);
}

// A fragment: 16x32 bf16, rows striped per lane (ISA 7.12.2 16-bit A layout).
// src layout: row-major [row][k] in LDS, bf16 units, even strides.
__device__ __forceinline__ v16bf load_a_frag(const unsigned short* __restrict__ base,
                                             int row0, int kbase, int stride, int lane) {
    const int half = lane >> 4, r = lane & 15;
    union { v16bf v; unsigned int u[8]; } f;
    const unsigned short* rp = base + (row0 + r) * stride + kbase + 8 * half;
#pragma unroll
    for (int p = 0; p < 8; ++p) {
        const int k = 2 * (p & 3) + 16 * (p >> 2);
        f.u[p] = *(const unsigned int*)(rp + k);
    }
    return f.v;
}

// B fragment: 32x16 bf16, one column per lane (ISA 7.12.2 / sparse B layout).
// src layout: [n][k] in LDS (k contiguous), bf16 units.
__device__ __forceinline__ v16bf load_b_frag(const unsigned short* __restrict__ base,
                                             int n0, int kbase, int stride, int lane) {
    const int half = lane >> 4, r = lane & 15;
    union { v16bf v; unsigned int u[8]; } f;
    const unsigned short* np = base + (n0 + r) * stride + kbase + 8 * half;
#pragma unroll
    for (int p = 0; p < 8; ++p) f.u[p] = *(const unsigned int*)(np + 2 * (p & 3) + 16 * (p >> 2));
    return f.v;
}

__global__ void __launch_bounds__(256)
gat_edge_kernel(const float* __restrict__ x_i, const float* __restrict__ x_j,
                const float* __restrict__ edge_attr,
                const float* __restrict__ W_in, const float* __restrict__ b_in,
                const float* __restrict__ ln_g, const float* __restrict__ ln_b,
                const float* __restrict__ W_alpha, const float* __restrict__ b_alpha,
                const float* __restrict__ W_mess, const float* __restrict__ b_mess,
                float* __restrict__ out_m, float* __restrict__ a_out, int E) {
    __shared__ __align__(16) unsigned short sWin[E_HIDDEN * WIN_STRIDE];   // [n][k] bf16
    __shared__ __align__(16) unsigned short sWm [E_NODE_DIM * HW_STRIDE];  // [n][k] bf16
    __shared__ __align__(16) unsigned short sA  [TILE_E * WIN_STRIDE];     // [edge][k] bf16
    __shared__ __align__(16) unsigned short sLkm[TILE_E * HW_STRIDE];      // leaky(mess) bf16
    __shared__ __align__(16) float sMess[TILE_E * HW_STRIDE];              // f32
    __shared__ float sWa[E_HIDDEN], sBin[E_HIDDEN], sG[E_HIDDEN], sB[E_HIDDEN];
    __shared__ float sBm[E_NODE_DIM];
    __shared__ float sBa;

    const int tid  = threadIdx.x;
    const int wave = tid >> 5;
    const int lane = tid & 31;
    const int half = lane >> 4;
    const int r    = lane & 15;

    // ---- one-time weight preload (f32 -> bf16, k-contiguous per n) ----
    for (int idx = tid; idx < E_INDIM * E_HIDDEN; idx += 256) {
        const int k = idx / E_HIDDEN, n = idx % E_HIDDEN;
        sWin[n * WIN_STRIDE + k] = f2bf(W_in[idx]);
    }
    for (int idx = tid; idx < E_HIDDEN * E_NODE_DIM; idx += 256) {
        const int k = idx / E_NODE_DIM, n = idx % E_NODE_DIM;
        sWm[n * HW_STRIDE + k] = f2bf(W_mess[idx]);
    }
    for (int idx = tid; idx < E_HIDDEN; idx += 256) {
        sWa[idx] = W_alpha[idx]; sBin[idx] = b_in[idx];
        sG[idx] = ln_g[idx];     sB[idx]  = ln_b[idx];
    }
    if (tid < E_NODE_DIM) sBm[tid] = b_mess[tid];
    if (tid == 0) sBa = b_alpha[0];

    const int nTiles = (E + TILE_E - 1) / TILE_E;
    for (int tile = blockIdx.x; tile < nTiles; tile += gridDim.x) {
        const int base = tile * TILE_E;
        __syncthreads();   // protect LDS reuse across iterations + weight preload

        // ---- stage concat([x_j, x_i, edge_attr]) tile as bf16 ----
        for (int idx = tid; idx < TILE_E * E_INDIM; idx += 256) {
            const int er = idx / E_INDIM, f = idx % E_INDIM;
            const int e = base + er;
            float v = 0.f;
            if (e < E) {
                if (f < E_NODE_DIM)            v = x_j[e * E_NODE_DIM + f];
                else if (f < 2 * E_NODE_DIM)   v = x_i[e * E_NODE_DIM + (f - E_NODE_DIM)];
                else                           v = edge_attr[e * E_EDGE_DIM + (f - 2 * E_NODE_DIM)];
            }
            sA[er * WIN_STRIDE + f] = f2bf(v);
        }
        __syncthreads();

        // ---- GEMM1: [32,160] x [160,192] via bf16 WMMA, 24 tiles / 8 waves ----
#pragma unroll
        for (int i = 0; i < 3; ++i) {
            const int t  = wave + 8 * i;
            const int mt = t / 12, nt = t % 12;
            v8f acc = {};
#pragma unroll
            for (int kk = 0; kk < 5; ++kk) {
                v16bf af = load_a_frag(sA,  mt * 16, kk * 32, WIN_STRIDE, lane);
                v16bf bf = load_b_frag(sWin, nt * 16, kk * 32, WIN_STRIDE, lane);
                acc = __builtin_amdgcn_wmma_f32_16x16x32_bf16(
                        false, af, false, bf, (short)0, acc, false, false);
            }
            const float bias = sBin[nt * 16 + r];
#pragma unroll
            for (int v = 0; v < 8; ++v)
                sMess[(mt * 16 + v + 8 * half) * HW_STRIDE + nt * 16 + r] = acc[v] + bias;
        }
        __syncthreads();

        // ---- LayerNorm + leaky + alpha logit (8 threads per edge row) ----
        {
            const int row = tid >> 3, sub = tid & 7;
            float s = 0.f, ss = 0.f;
#pragma unroll
            for (int j = 0; j < 24; ++j) {
                const float v = sMess[row * HW_STRIDE + sub + 8 * j];
                s += v; ss += v * v;
            }
#pragma unroll
            for (int m = 1; m <= 4; m <<= 1) {
                s  += __shfl_xor(s,  m, 32);
                ss += __shfl_xor(ss, m, 32);
            }
            const float mu   = s * (1.f / 192.f);
            const float var  = ss * (1.f / 192.f) - mu * mu;
            const float rstd = rsqrtf(var + 1e-5f);
            float aacc = 0.f;
#pragma unroll
            for (int j = 0; j < 24; ++j) {
                const int c = sub + 8 * j;
                const float v  = sMess[row * HW_STRIDE + c];
                const float ln = (v - mu) * rstd * sG[c] + sB[c];
                aacc += (ln >= 0.f ? ln : 0.01f * ln) * sWa[c];
                const float lk = (v >= 0.f ? v : 0.01f * v);
                sLkm[row * HW_STRIDE + c] = f2bf(lk);
            }
#pragma unroll
            for (int m = 1; m <= 4; m <<= 1) aacc += __shfl_xor(aacc, m, 32);
            if (sub == 0) {
                const int e = base + row;
                if (e < E) a_out[e] = aacc + sBa;
            }
        }
        __syncthreads();

        // ---- GEMM2: [32,192] x [192,64] via bf16 WMMA, 8 tiles / 8 waves ----
        {
            const int mt = wave >> 2, nt = wave & 3;
            v8f acc = {};
#pragma unroll
            for (int kk = 0; kk < 6; ++kk) {
                v16bf af = load_a_frag(sLkm, mt * 16, kk * 32, HW_STRIDE, lane);
                v16bf bf = load_b_frag(sWm,  nt * 16, kk * 32, HW_STRIDE, lane);
                acc = __builtin_amdgcn_wmma_f32_16x16x32_bf16(
                        false, af, false, bf, (short)0, acc, false, false);
            }
            const int n = nt * 16 + r;
            const float bias = sBm[n];
#pragma unroll
            for (int v = 0; v < 8; ++v) {
                const int e = base + mt * 16 + v + 8 * half;
                if (e < E) out_m[e * E_NODE_DIM + n] = acc[v] + bias;
            }
        }
    }
}

__global__ void seg_init_kernel(float* __restrict__ smax, float* __restrict__ ssum, int n) {
    const int i = blockIdx.x * blockDim.x + threadIdx.x;
    if (i < n) {
        smax[i] = __uint_as_float(0xFF800000u);   // -inf
        ssum[i] = 0.f;
    }
}

__device__ __forceinline__ void atomicMaxF(float* addr, float val) {
    if (val >= 0.f) atomicMax((int*)addr, __float_as_int(val));
    else            atomicMin((unsigned int*)addr, __float_as_uint(val));
}

__global__ void seg_max_kernel(const float* __restrict__ a, const int* __restrict__ col,
                               float* __restrict__ smax, int E) {
    const int e = blockIdx.x * blockDim.x + threadIdx.x;
    if (e < E) atomicMaxF(&smax[col[e]], a[e]);
}

__global__ void seg_exp_sum_kernel(float* __restrict__ a, const int* __restrict__ col,
                                   const float* __restrict__ smax, float* __restrict__ ssum, int E) {
    const int e = blockIdx.x * blockDim.x + threadIdx.x;
    if (e < E) {
        const int c = col[e];
        const float ex = __expf(a[e] - smax[c]);
        a[e] = ex;                     // reuse logits buffer for exp values
        atomicAdd(&ssum[c], ex);
    }
}

__global__ void scale_out_kernel(float* __restrict__ out, const float* __restrict__ ex,
                                 const int* __restrict__ col, const float* __restrict__ ssum,
                                 int E) {
    const int i = blockIdx.x * blockDim.x + threadIdx.x;
    if (i < E * E_NODE_DIM) {
        const int e = i >> 6;          // / 64
        const float alpha = ex[e] / (ssum[col[e]] + 1e-16f);
        out[i] *= alpha;
    }
}

extern "C" void kernel_launch(void* const* d_in, const int* in_sizes, int n_in,
                              void* d_out, int out_size, void* d_ws, size_t ws_size,
                              hipStream_t stream) {
    (void)n_in; (void)out_size; (void)ws_size;
    const float* x_i       = (const float*)d_in[0];
    const float* x_j       = (const float*)d_in[1];
    const float* edge_attr = (const float*)d_in[2];
    const float* W_in      = (const float*)d_in[3];
    const float* b_in      = (const float*)d_in[4];
    const float* ln_g      = (const float*)d_in[5];
    const float* ln_b      = (const float*)d_in[6];
    const float* W_alpha   = (const float*)d_in[7];
    const float* b_alpha   = (const float*)d_in[8];
    const float* W_mess    = (const float*)d_in[9];
    const float* b_mess    = (const float*)d_in[10];
    const int*   eidx      = (const int*)d_in[11];
    const int E = in_sizes[0] / E_NODE_DIM;
    const int* col = eidx + E;                       // edge_index[1]

    float* ws   = (float*)d_ws;
    float* aE   = ws;                                 // E floats: logits -> exp values
    float* smax = ws + E;                             // NUM_NODES floats
    float* ssum = smax + E_NNODES;                    // NUM_NODES floats
    float* outm = (float*)d_out;

    seg_init_kernel<<<(E_NNODES + 255) / 256, 256, 0, stream>>>(smax, ssum, E_NNODES);

    const int nTiles = (E + TILE_E - 1) / TILE_E;
    const int gridA  = nTiles < 4096 ? nTiles : 4096;
    gat_edge_kernel<<<gridA, 256, 0, stream>>>(x_i, x_j, edge_attr, W_in, b_in,
                                               ln_g, ln_b, W_alpha, b_alpha,
                                               W_mess, b_mess, outm, aE, E);

    seg_max_kernel<<<(E + 255) / 256, 256, 0, stream>>>(aE, col, smax, E);
    seg_exp_sum_kernel<<<(E + 255) / 256, 256, 0, stream>>>(aE, col, smax, ssum, E);
    scale_out_kernel<<<((long long)E * E_NODE_DIM + 255) / 256, 256, 0, stream>>>(
        outm, aE, col, ssum, E);
}
